// net_jknet_84524956385823
// MI455X (gfx1250) — matrix operand order repaired and checked by reading, not compiled, not run on hardware
//
#include <hip/hip_runtime.h>
#include <math.h>

// ---------------------------------------------------------------------------
// MI455X / gfx1250 implementation.
// Dense matmuls use v_wmma_f32_16x16x32_f16 (wave32, f32 accumulate).
// B (weight) matrices are pre-packed into the WMMA B-fragment register
// layout (CDNA5 ISA 7.12.2) so each fragment is a single
// contiguous 32-byte per-lane load (2x global_load_b128, coalesced):
//   fragment tile (kt, nt): Bp[(nt*ktiles + kt)*512 + lane*16 + t]
//     = W[kt*32 + (lane>=16?8:0) + (t<8 ? t : t+8)][nt*16 + (lane&15)]
// A fragments load directly from row-major f16 (2x b128 per fragment):
//   lane<16 -> row=lane,    K = {k0..k0+7, k0+16..k0+23}
//   lane>=16 -> row=lane-16, K shifted by +8
// C/D: vgpr r -> row = r + (lane>=16 ? 8 : 0), col = lane&15
// ---------------------------------------------------------------------------

typedef __attribute__((ext_vector_type(16))) _Float16 v16h;
typedef __attribute__((ext_vector_type(8)))  _Float16 v8h;
typedef __attribute__((ext_vector_type(8)))  float    v8f;

#define NNODES 50000
#define NEDGES 800000

// ---------------------------- small utility kernels ------------------------

__global__ void f32_to_f16_kernel(const float* __restrict__ s,
                                  _Float16* __restrict__ d, int n) {
  int i = blockIdx.x * blockDim.x + threadIdx.x;
  if (i < n) d[i] = (_Float16)s[i];
}

// Pack W[K,Nc] (f32 row-major) into WMMA B-fragment layout (f16).
__global__ void pack_b_kernel(const float* __restrict__ W,
                              _Float16* __restrict__ Bp, int K, int Nc) {
  int idx = blockIdx.x * blockDim.x + threadIdx.x;
  if (idx >= K * Nc) return;
  int within = idx & 511;       // element within a 32x16 tile (512 elems)
  int tile   = idx >> 9;
  int lane   = within >> 4;
  int t      = within & 15;
  int hi     = lane >> 4;
  int colL   = lane & 15;
  int ktiles = K >> 5;
  int nt = tile / ktiles;
  int kt = tile - nt * ktiles;
  int srcK = kt * 32 + (t < 8 ? hi * 8 + t : 16 + hi * 8 + (t - 8));
  int srcN = nt * 16 + colL;
  Bp[idx] = (_Float16)W[(size_t)srcK * Nc + srcN];
}

__global__ void parsing_kernel(const float* __restrict__ p0,
                               float* __restrict__ P, int n) {
  int i = blockIdx.x * blockDim.x + threadIdx.x;
  if (i < n) P[i] = fmaxf(2.0f * p0[i], 0.0f);
}

// deg[i] = 1 (self loop weight), stats zeroed
__global__ void init_kernel(float* __restrict__ deg, float* __restrict__ stats,
                            int n) {
  int i = blockIdx.x * blockDim.x + threadIdx.x;
  if (i < 8) stats[i] = 0.0f;
  if (i < n) deg[i] = 1.0f;
}

__global__ void rsqrt_kernel(float* __restrict__ deg, int n) {
  int i = blockIdx.x * blockDim.x + threadIdx.x;
  if (i < n) {
    float d = deg[i];
    deg[i] = (d > 0.0f) ? rsqrtf(d) : 0.0f;
  }
}

// ------------------------------- WMMA GEMM ---------------------------------
// C[M,Nc] = A[M,K](f16,row-major,lda) * Bp(packed) (+bias)(relu?).
// Each wave: 32 rows x 64 cols. Requires Nc%64==0, K%32==0, M%16==0
// (all true here; M%32 may be 16 -> tail handled by clamp + guarded store).
__global__ void gemm_wmma_kernel(const _Float16* __restrict__ A, int lda,
                                 const _Float16* __restrict__ Bp,
                                 const float* __restrict__ bias, int relu,
                                 float* __restrict__ Cf, int ldcf,
                                 _Float16* __restrict__ Ch, int ldch,
                                 int M, int Nc, int K) {
  int t      = blockIdx.x * blockDim.x + threadIdx.x;
  int wv     = t >> 5;
  int lane   = t & 31;
  int nstrip = Nc >> 6;
  int m0 = (wv / nstrip) * 32;
  int n0 = (wv % nstrip) * 64;
  if (m0 >= M) return;

  int rlane  = lane & 15;
  int hi     = lane >> 4;
  int ktiles = K >> 5;

  int rowA0 = m0 + rlane;
  int rowA1 = m0 + 16 + rlane;
  if (rowA1 >= M) rowA1 = M - 1;   // tail clamp (in-bounds read, discarded)

  v8f acc[8] = {};

  const _Float16* A0 = A + (size_t)rowA0 * lda + (size_t)hi * 8;
  const _Float16* A1 = A + (size_t)rowA1 * lda + (size_t)hi * 8;
  const _Float16* Bl = Bp + (size_t)(n0 >> 4) * ktiles * 512 + (size_t)lane * 16;

  for (int kt = 0; kt < ktiles; ++kt) {
    int k0 = kt * 32;
    v8h a0lo = *(const v8h*)(A0 + k0);
    v8h a0hi = *(const v8h*)(A0 + k0 + 16);
    v8h a1lo = *(const v8h*)(A1 + k0);
    v8h a1hi = *(const v8h*)(A1 + k0 + 16);
    v16h a0, a1;
#pragma unroll
    for (int i = 0; i < 8; ++i) {
      a0[i] = a0lo[i]; a0[i + 8] = a0hi[i];
      a1[i] = a1lo[i]; a1[i + 8] = a1hi[i];
    }
#pragma unroll
    for (int j = 0; j < 4; ++j) {
      v16h b = *(const v16h*)(Bl + ((size_t)j * ktiles + kt) * 512);
      acc[j]     = __builtin_amdgcn_wmma_f32_16x16x32_f16(
          false, a0, false, b, (short)0, acc[j], false, false);
      acc[4 + j] = __builtin_amdgcn_wmma_f32_16x16x32_f16(
          false, a1, false, b, (short)0, acc[4 + j], false, false);
    }
  }

  int rb = m0 + hi * 8;
#pragma unroll
  for (int j = 0; j < 4; ++j) {
    int cc = n0 + j * 16 + rlane;
    float bv = bias ? bias[cc] : 0.0f;
#pragma unroll
    for (int r = 0; r < 8; ++r) {
      float v = acc[j][r] + bv;
      if (relu) v = fmaxf(v, 0.0f);
      size_t ro = (size_t)(rb + r);
      if (Cf) Cf[ro * (size_t)ldcf + cc] = v;
      if (Ch) Ch[ro * (size_t)ldch + cc] = (_Float16)v;
    }
#pragma unroll
    for (int r = 0; r < 8; ++r) {
      int row = rb + 16 + r;
      if (row < M) {
        float v = acc[4 + j][r] + bv;
        if (relu) v = fmaxf(v, 0.0f);
        size_t ro = (size_t)row;
        if (Cf) Cf[ro * (size_t)ldcf + cc] = v;
        if (Ch) Ch[ro * (size_t)ldch + cc] = (_Float16)v;
      }
    }
  }
}

// -------------------- edge-weight pipeline (soft edges) --------------------

// t[i,k] = sum_d logits[i,d] * P[d,k]   (P cached in LDS, 64x64 f32)
__global__ void tmat_kernel(const float* __restrict__ logits,
                            const float* __restrict__ P,
                            float* __restrict__ tmat, int total) {
  __shared__ float sp[64 * 64];
  for (int i = threadIdx.x; i < 64 * 64; i += blockDim.x) sp[i] = P[i];
  __syncthreads();
  int idx = blockIdx.x * blockDim.x + threadIdx.x;
  if (idx >= total) return;
  int i = idx >> 6, k = idx & 63;
  const float* L = logits + (size_t)i * 64;
  float acc = 0.0f;
#pragma unroll
  for (int d = 0; d < 64; ++d) acc += L[d] * sp[d * 64 + k];
  tmat[idx] = acc;
}

// ew_raw[e] = dot(logits[row[e]], t[col[e]]); accumulate sum / sumsq.
__global__ void edge_ew_kernel(const int* __restrict__ row,
                               const int* __restrict__ col,
                               const float* __restrict__ logits,
                               const float* __restrict__ tmat,
                               float* __restrict__ ew,
                               float* __restrict__ stats, int E) {
  __shared__ float s1[256];
  __shared__ float s2[256];
  int e = blockIdx.x * blockDim.x + threadIdx.x;
  float d = 0.0f;
  if (e < E) {
    int r = row[e], c = col[e];
    const float4* sp = (const float4*)(logits + (size_t)r * 64);
    const float4* tp = (const float4*)(tmat + (size_t)c * 64);
#pragma unroll
    for (int i = 0; i < 16; ++i) {
      float4 a = sp[i], b = tp[i];
      d += a.x * b.x + a.y * b.y + a.z * b.z + a.w * b.w;
    }
    ew[e] = d;
  }
  s1[threadIdx.x] = d;
  s2[threadIdx.x] = d * d;
  __syncthreads();
  for (int s = 128; s > 0; s >>= 1) {
    if ((int)threadIdx.x < s) {
      s1[threadIdx.x] += s1[threadIdx.x + s];
      s2[threadIdx.x] += s2[threadIdx.x + s];
    }
    __syncthreads();
  }
  if (threadIdx.x == 0) {
    atomicAdd(&stats[0], s1[0]);
    atomicAdd(&stats[1], s2[0]);
  }
}

// ew = (ew - mean) * sqrt(1e-4 / var_unbiased) + 1
__global__ void ew_norm_kernel(float* __restrict__ ew,
                               const float* __restrict__ stats, int E) {
  int e = blockIdx.x * blockDim.x + threadIdx.x;
  if (e >= E) return;
  float sum = stats[0], sq = stats[1];
  float fe = (float)E;
  float mean = sum / fe;
  float var = (sq - sum * sum / fe) / (fe - 1.0f);
  float scale = sqrtf(1e-4f / var);
  ew[e] = (ew[e] - mean) * scale + 1.0f;
}

__global__ void edge_deg_kernel(const int* __restrict__ col,
                                const float* __restrict__ ew,
                                float* __restrict__ deg, int E) {
  int e = blockIdx.x * blockDim.x + threadIdx.x;
  if (e < E) atomicAdd(&deg[col[e]], ew[e]);
}

// -------------------------- GCN aggregation (F=128) ------------------------

// agg[i,:] = dinv[i]^2 * xw[i,:]   (self loop, weight 1)
__global__ void self_init_kernel(const float* __restrict__ dinv,
                                 const float* __restrict__ xw,
                                 float* __restrict__ agg, int total) {
  int idx = blockIdx.x * blockDim.x + threadIdx.x;
  if (idx >= total) return;
  int i = idx >> 7;
  float di = dinv[i];
  agg[idx] = di * di * xw[idx];
}

// one wave per edge, each lane handles 4 of 128 features
__global__ void scatter_kernel(const int* __restrict__ row,
                               const int* __restrict__ col,
                               const float* __restrict__ ew,
                               const float* __restrict__ dinv,
                               const float* __restrict__ xw,
                               float* __restrict__ agg, int E) {
  int t = blockIdx.x * blockDim.x + threadIdx.x;
  int wv = t >> 5;
  int lane = t & 31;
  if (wv >= E) return;
  int r = row[wv], c = col[wv];
  float nrm = dinv[r] * ew[wv] * dinv[c];
  float4 v = *(const float4*)(xw + (size_t)r * 128 + lane * 4);
  float* dst = agg + (size_t)c * 128 + lane * 4;
  atomicAdd(dst + 0, nrm * v.x);
  atomicAdd(dst + 1, nrm * v.y);
  atomicAdd(dst + 2, nrm * v.z);
  atomicAdd(dst + 3, nrm * v.w);
}

// dst[i*ldd + f] = f16(relu(agg[i*128+f] + bias[f]))  (writes into xj slot)
__global__ void bias_relu_f16_kernel(const float* __restrict__ agg,
                                     const float* __restrict__ bias,
                                     _Float16* __restrict__ dst, int ldd,
                                     int total) {
  int idx = blockIdx.x * blockDim.x + threadIdx.x;
  if (idx >= total) return;
  int i = idx >> 7, f = idx & 127;
  float v = fmaxf(agg[idx] + bias[f], 0.0f);
  dst[(size_t)i * ldd + f] = (_Float16)v;
}

// ------------------------------- launcher ----------------------------------

static inline void launch_gemm(hipStream_t stream, const _Float16* A, int lda,
                               const _Float16* Bp, const float* bias, int relu,
                               float* Cf, int ldcf, _Float16* Ch, int ldch,
                               int M, int Nc, int K) {
  long mt = (M + 31) / 32;
  long waves = mt * (Nc >> 6);
  long thr = waves * 32;
  int blocks = (int)((thr + 255) / 256);
  gemm_wmma_kernel<<<blocks, 256, 0, stream>>>(A, lda, Bp, bias, relu, Cf,
                                               ldcf, Ch, ldch, M, Nc, K);
}

extern "C" void kernel_launch(void* const* d_in, const int* in_sizes, int n_in,
                              void* d_out, int out_size, void* d_ws,
                              size_t ws_size, hipStream_t stream) {
  (void)in_sizes; (void)n_in; (void)out_size; (void)ws_size;
  const int N = NNODES, E = NEDGES;

  // ---------------- inputs (setup_inputs order) ----------------
  const float* x    = (const float*)d_in[0];    // [N,512]
  const int*   ei   = (const int*)d_in[1];      // [2,E]
  const float* Wp1  = (const float*)d_in[2];    // [512,512]
  const float* bp1  = (const float*)d_in[3];
  const float* Wp2  = (const float*)d_in[4];    // [512,64]
  const float* bp2  = (const float*)d_in[5];
  const float* Wp3  = (const float*)d_in[6];    // [64,64]
  const float* bp3  = (const float*)d_in[7];
  const float* par0 = (const float*)d_in[8];    // [64,64]
  const float* Wg0  = (const float*)d_in[9];    // [512,128]
  const float* bg0  = (const float*)d_in[10];
  const float* Wg1  = (const float*)d_in[11];   // [128,128]
  const float* bg1  = (const float*)d_in[12];
  // d_in[13], d_in[14] (Wg2,bg2): dead in the reference dataflow
  const float* Wl1  = (const float*)d_in[15];   // [256,128]
  const float* bl1  = (const float*)d_in[16];
  const float* Wl2  = (const float*)d_in[17];   // [128,64]
  const float* bl2  = (const float*)d_in[18];
  float* out = (float*)d_out;                   // [N,64]

  const int* erow = ei;
  const int* ecol = ei + E;

  // ---------------- workspace layout (liveness-overlapped) ----------------
  char* w = (char*)d_ws;
  size_t o = 0;
  auto take = [&](size_t bytes) -> char* {
    char* p = w + o;
    o += (bytes + 255) & ~(size_t)255;
    return p;
  };
  _Float16* xh = (_Float16*)take((size_t)N * 512 * 2);  // x in f16
  char* r1 = take((size_t)N * 512 * 2);  // h1h  OR  xw(0..)+agg(+25.6MB)
  char* r2 = take((size_t)N * 128 * 2);  // h2h  OR  hid
  char* r3 = take((size_t)N * 256 * 2);  // logits+tmat  OR  xj
  _Float16* Wp1h = (_Float16*)take(512 * 512 * 2);
  _Float16* Wp2h = (_Float16*)take(512 * 64 * 2);
  _Float16* Wp3h = (_Float16*)take(64 * 64 * 2);
  _Float16* Wg0h = (_Float16*)take(512 * 128 * 2);
  _Float16* Wg1h = (_Float16*)take(128 * 128 * 2);
  _Float16* Wl1h = (_Float16*)take(256 * 128 * 2);
  _Float16* Wl2h = (_Float16*)take(128 * 64 * 2);
  float* Pf    = (float*)take(64 * 64 * 4);
  float* ew    = (float*)take((size_t)E * 4);
  float* stats = (float*)take(256);
  float* dinv  = (float*)take((size_t)N * 4);   // deg, then rsqrt in place

  _Float16* h1h   = (_Float16*)r1;
  float*    xw    = (float*)r1;
  float*    agg   = (float*)(r1 + (size_t)N * 128 * 4);
  _Float16* h2h   = (_Float16*)r2;
  _Float16* hid   = (_Float16*)r2;
  float*    logit = (float*)r3;
  float*    tmat  = (float*)(r3 + (size_t)N * 64 * 4);
  _Float16* xj    = (_Float16*)r3;              // [N,256] f16 JK concat

  auto grid = [](long n) { return (int)((n + 255) / 256); };

  // ---------------- 1) convert x; pack weights into B-fragment layout ------
  f32_to_f16_kernel<<<grid((long)N * 512), 256, 0, stream>>>(x, xh, N * 512);
  pack_b_kernel<<<grid(512 * 512), 256, 0, stream>>>(Wp1, Wp1h, 512, 512);
  pack_b_kernel<<<grid(512 * 64), 256, 0, stream>>>(Wp2, Wp2h, 512, 64);
  pack_b_kernel<<<grid(64 * 64), 256, 0, stream>>>(Wp3, Wp3h, 64, 64);
  pack_b_kernel<<<grid(512 * 128), 256, 0, stream>>>(Wg0, Wg0h, 512, 128);
  pack_b_kernel<<<grid(128 * 128), 256, 0, stream>>>(Wg1, Wg1h, 128, 128);
  pack_b_kernel<<<grid(256 * 128), 256, 0, stream>>>(Wl1, Wl1h, 256, 128);
  pack_b_kernel<<<grid(128 * 64), 256, 0, stream>>>(Wl2, Wl2h, 128, 64);
  parsing_kernel<<<grid(64 * 64), 256, 0, stream>>>(par0, Pf, 64 * 64);

  // ---------------- 2) pseudo MLP (WMMA) ----------------
  launch_gemm(stream, xh, 512, Wp1h, bp1, 1, nullptr, 0, h1h, 512, N, 512, 512);
  launch_gemm(stream, h1h, 512, Wp2h, bp2, 1, nullptr, 0, h2h, 64, N, 64, 512);
  launch_gemm(stream, h2h, 64, Wp3h, bp3, 0, logit, 64, nullptr, 0, N, 64, 64);

  // ---------------- 3) soft edge weights ----------------
  tmat_kernel<<<grid((long)N * 64), 256, 0, stream>>>(logit, Pf, tmat, N * 64);
  init_kernel<<<grid(N), 256, 0, stream>>>(dinv, stats, N);
  edge_ew_kernel<<<grid(E), 256, 0, stream>>>(erow, ecol, logit, tmat, ew, stats, E);
  ew_norm_kernel<<<grid(E), 256, 0, stream>>>(ew, stats, E);
  edge_deg_kernel<<<grid(E), 256, 0, stream>>>(ecol, ew, dinv, E);
  rsqrt_kernel<<<grid(N), 256, 0, stream>>>(dinv, N);

  // ---------------- 4) GCN layer 0 ----------------
  launch_gemm(stream, xh, 512, Wg0h, nullptr, 0, xw, 128, nullptr, 0, N, 128, 512);
  self_init_kernel<<<grid((long)N * 128), 256, 0, stream>>>(dinv, xw, agg, N * 128);
  scatter_kernel<<<grid((long)E * 32), 256, 0, stream>>>(erow, ecol, ew, dinv, xw, agg, E);
  bias_relu_f16_kernel<<<grid((long)N * 128), 256, 0, stream>>>(agg, bg0, xj, 256, N * 128);

  // ---------------- 5) GCN layer 1 (reads x1 from xj, lda=256) -------------
  launch_gemm(stream, xj, 256, Wg1h, nullptr, 0, xw, 128, nullptr, 0, N, 128, 128);
  self_init_kernel<<<grid((long)N * 128), 256, 0, stream>>>(dinv, xw, agg, N * 128);
  scatter_kernel<<<grid((long)E * 32), 256, 0, stream>>>(erow, ecol, ew, dinv, xw, agg, E);
  bias_relu_f16_kernel<<<grid((long)N * 128), 256, 0, stream>>>(agg, bg1, xj + 128, 256, N * 128);

  // ---------------- 6) JK head: relu(xj@Wl1+bl1) @ Wl2 + bl2 ---------------
  launch_gemm(stream, xj, 256, Wl1h, bl1, 1, nullptr, 0, hid, 128, N, 128, 256);
  launch_gemm(stream, hid, 128, Wl2h, bl2, 0, out, 64, nullptr, 0, N, 64, 128);
}
